// ChamferLoss_592705487396
// MI455X (gfx1250) — matrix operand order repaired and checked by reading, not compiled, not run on hardware
//
#include <hip/hip_runtime.h>

typedef __attribute__((ext_vector_type(16))) _Float16 v16h;
typedef __attribute__((ext_vector_type(8)))  float    v8f;

constexpr int Bn = 512;
constexpr int Kn = 1024;
constexpr int Mn = 64;
constexpr int Dn = 128;
constexpr int ROWH = 136;   // padded LDS row stride in halves (272B, 16B aligned, bank-spread)

// 1/max(sqrt(s), 1e-12)  ==  rsq(max(s, 1e-24)) : single v_rsq_f32
__device__ __forceinline__ float fast_inv_norm(float s) {
    return __builtin_amdgcn_rsqf(fmaxf(s, 1e-24f));
}

__global__ __launch_bounds__(256, 4)   // target >=4 waves/SIMD
void chamfer_main(const float* __restrict__ tokens,
                  const float* __restrict__ interests,
                  float* __restrict__ wsI,   // [Bn] per-batch sum of per-interest min dist
                  float* __restrict__ wsT)   // [Bn] per-batch sum of per-token   min dist
{
    __shared__ _Float16 sI[Mn * ROWH];   // normalized interests, f16
    __shared__ float    sI2[Mn];         // ||i_hat||^2 per interest
    __shared__ unsigned sMinI[Mn];       // per-interest min distance (float bits, >=0)
    __shared__ float    sSumT;           // block sum of per-token mins

    const int b   = blockIdx.x;
    const int tid = threadIdx.x;

    // ---------------- Phase 1: normalize interests into LDS (f16) ----------------
    {
        const int m  = tid >> 2;          // 4 threads per interest row
        const int q  = tid & 3;
        const int d0 = q * 32;
        const float* ip = interests + ((size_t)b * Mn + m) * Dn + d0;
        float v[32];
        float s = 0.f;
        #pragma unroll
        for (int j = 0; j < 8; ++j) {
            float4 x = ((const float4*)ip)[j];
            v[4*j+0] = x.x; v[4*j+1] = x.y; v[4*j+2] = x.z; v[4*j+3] = x.w;
            s += x.x*x.x + x.y*x.y + x.z*x.z + x.w*x.w;
        }
        s += __shfl_xor(s, 1);
        s += __shfl_xor(s, 2);
        const float scale = fast_inv_norm(s);
        #pragma unroll
        for (int j = 0; j < 32; ++j)
            sI[m * ROWH + d0 + j] = (_Float16)(v[j] * scale);
        if (q == 0) sI2[m] = s * scale * scale;
    }
    if (tid < Mn) sMinI[tid] = 0x7f800000u;  // +inf bits
    if (tid == 0) sSumT = 0.f;
    __syncthreads();

    // ---------------- Phase 2: stream token tiles, WMMA dots, reductions ----------------
    const int wave = tid >> 5;
    const int lane = tid & 31;
    const int l16  = lane & 15;
    const int hi   = lane >> 4;
    const int offA = hi ? 8  : 0;   // D-row offset of this lane half (acc rows 8*hi + v)
    const int offB = hi ? 16 : 0;   // K offset of this lane half for A and B (contiguous k = 16h + s)

    float i2v[4];
    #pragma unroll
    for (int nt = 0; nt < 4; ++nt) i2v[nt] = sI2[nt * 16 + l16];

    float runMin[4] = { __builtin_inff(), __builtin_inff(), __builtin_inff(), __builtin_inff() };
    float sumT_local = 0.f;

    for (int kt = wave; kt < Kn / 16; kt += 8) {
        const int row = kt * 16 + l16;
        const float* tp = tokens + ((size_t)b * Kn + row) * Dn;

        // Load this lane's share of the 16x128 token tile (A-layout chunks).
        // Convert RAW values to f16 immediately (fp32 staging dies per-chunk);
        // the row normalization scale is applied to the WMMA result instead:
        //   dot(t_hat, i_hat) = scale_row * dot(t_raw_f16, i_hat)
        v16h afrag[4];
        float s = 0.f;
        #pragma unroll
        for (int c = 0; c < 4; ++c) {
            #pragma unroll
            for (int h = 0; h < 2; ++h) {
                const int d = 32*c + offB + 8*h;
                float4 x0 = *(const float4*)(tp + d);
                float4 x1 = *(const float4*)(tp + d + 4);
                s += x0.x*x0.x + x0.y*x0.y + x0.z*x0.z + x0.w*x0.w
                   + x1.x*x1.x + x1.y*x1.y + x1.z*x1.z + x1.w*x1.w;
                const int o = h*8;
                afrag[c][o+0] = (_Float16)x0.x; afrag[c][o+1] = (_Float16)x0.y;
                afrag[c][o+2] = (_Float16)x0.z; afrag[c][o+3] = (_Float16)x0.w;
                afrag[c][o+4] = (_Float16)x1.x; afrag[c][o+5] = (_Float16)x1.y;
                afrag[c][o+6] = (_Float16)x1.z; afrag[c][o+7] = (_Float16)x1.w;
            }
        }
        s += __shfl_xor(s, 16);                  // full row sumsq (two lanes per row)
        const float scale = fast_inv_norm(s);    // v_rsq_f32
        const float t2    = s * scale * scale;

        // per-output-row broadcast: row (v + hi*8) data lives in lane (v + offA)
        float t2row[8], scrow[8];
        #pragma unroll
        for (int v = 0; v < 8; ++v) {
            t2row[v] = __shfl(t2,    v + offA);
            scrow[v] = __shfl(scale, v + offA);
        }

        float kmin[8];
        #pragma unroll
        for (int v = 0; v < 8; ++v) kmin[v] = __builtin_inff();

        #pragma unroll
        for (int nt = 0; nt < 4; ++nt) {
            v8f acc = {0.f,0.f,0.f,0.f,0.f,0.f,0.f,0.f};
            const int mrow = nt * 16 + l16;
            #pragma unroll
            for (int c = 0; c < 4; ++c) {
                const int baseh = mrow * ROWH + offB + 32*c;
                v16h bf;
                #pragma unroll
                for (int j = 0; j < 16; ++j) bf[j] = sI[baseh + j];
                acc = __builtin_amdgcn_wmma_f32_16x16x32_f16(
                        false, afrag[c], false, bf, (short)0, acc, false, false);
            }
            float mmin = __builtin_inff();
            #pragma unroll
            for (int v = 0; v < 8; ++v) {
                const float d2 = t2row[v] + i2v[nt] - 2.f * (scrow[v] * acc[v]);
                const float dd = __builtin_amdgcn_sqrtf(fmaxf(d2, 1e-12f)); // raw v_sqrt_f32
                kmin[v] = fminf(kmin[v], dd);
                mmin    = fminf(mmin, dd);
            }
            runMin[nt] = fminf(runMin[nt], mmin);
        }

        // token -> nearest interest: reduce over the 16 column-lanes, sum over rows
        float ksum = 0.f;
        #pragma unroll
        for (int v = 0; v < 8; ++v) {
            float kv = kmin[v];
            kv = fminf(kv, __shfl_xor(kv, 1));
            kv = fminf(kv, __shfl_xor(kv, 2));
            kv = fminf(kv, __shfl_xor(kv, 4));
            kv = fminf(kv, __shfl_xor(kv, 8));
            ksum += kv;
        }
        if (l16 == 0) sumT_local += ksum;   // lanes 0 (rows 0-7) and 16 (rows 8-15)
    }

    // interest -> nearest token: merge halves, LDS uint-min (distances >= 0)
    #pragma unroll
    for (int nt = 0; nt < 4; ++nt) {
        float r = fminf(runMin[nt], __shfl_xor(runMin[nt], 16));
        if (hi == 0) atomicMin(&sMinI[nt * 16 + l16], __float_as_uint(r));
    }
    sumT_local += __shfl_xor(sumT_local, 16);
    if (lane == 0) atomicAdd(&sSumT, sumT_local);

    __syncthreads();

    if (tid < 32) {
        float v = __uint_as_float(sMinI[tid]) + __uint_as_float(sMinI[tid + 32]);
        v += __shfl_xor(v, 1);
        v += __shfl_xor(v, 2);
        v += __shfl_xor(v, 4);
        v += __shfl_xor(v, 8);
        v += __shfl_xor(v, 16);
        if (tid == 0) wsI[b] = v;
    }
    if (tid == 0) wsT[b] = sSumT;
}

__global__ __launch_bounds__(512)
void chamfer_finalize(const float* __restrict__ wsI,
                      const float* __restrict__ wsT,
                      float* __restrict__ out)
{
    __shared__ float redI[16];
    __shared__ float redT[16];
    const int tid = threadIdx.x;
    float vI = wsI[tid];
    float vT = wsT[tid];
    #pragma unroll
    for (int m = 16; m > 0; m >>= 1) {
        vI += __shfl_xor(vI, m);
        vT += __shfl_xor(vT, m);
    }
    const int w = tid >> 5, l = tid & 31;
    if (l == 0) { redI[w] = vI; redT[w] = vT; }
    __syncthreads();
    if (tid == 0) {
        float sI = 0.f, sT = 0.f;
        #pragma unroll
        for (int i = 0; i < 16; ++i) { sI += redI[i]; sT += redT[i]; }
        out[0] = sI / (float)(Bn * Mn) + 0.3f * (sT / (float)(Bn * Kn));
    }
}

extern "C" void kernel_launch(void* const* d_in, const int* in_sizes, int n_in,
                              void* d_out, int out_size, void* d_ws, size_t ws_size,
                              hipStream_t stream)
{
    const float* tokens    = (const float*)d_in[0];
    const float* interests = (const float*)d_in[1];
    float* out = (float*)d_out;
    float* wsI = (float*)d_ws;        // Bn floats
    float* wsT = wsI + Bn;            // Bn floats

    chamfer_main<<<Bn, 256, 0, stream>>>(tokens, interests, wsI, wsT);
    chamfer_finalize<<<1, 512, 0, stream>>>(wsI, wsT, out);
}
